// MultichannelMultiheadAttention_25434796327665
// MI455X (gfx1250) — matrix operand order repaired and checked by reading, not compiled, not run on hardware
//
#include <hip/hip_runtime.h>
#include <hip/hip_bf16.h>

#define B_  2
#define C_  8
#define F_  512
#define W_  1024
#define H_  8
#define DH_ 64
#define BC_  (B_*C_)
#define BCH_ (B_*C_*H_)

typedef __attribute__((ext_vector_type(16))) _Float16 v16h;
typedef __attribute__((ext_vector_type(8)))  _Float16 v8h;
typedef __attribute__((ext_vector_type(2)))  _Float16 v2h;
typedef __attribute__((ext_vector_type(8)))  float    v8f;
typedef __attribute__((ext_vector_type(8)))  unsigned int v8u;

#define DEV static __device__ __forceinline__

DEV v16h cat8(v8h a, v8h b) {
  return __builtin_shufflevector(a, b, 0,1,2,3,4,5,6,7,8,9,10,11,12,13,14,15);
}

// A-fragment (16 M-rows x 32 K) from row-major storage, ld in halfs.
// Lane lr (=lane%16) holds row row0+lr; hl=0 lanes: K 0..15, hl=1: K 16..31.
DEV v16h load_a16x32(const _Float16* __restrict__ base, int row0, int k0, int ld, int lr, int hl) {
  const _Float16* p = base + (size_t)(row0 + lr) * ld + (k0 + 16 * hl);
  v8h lo = *(const v8h*)p;
  v8h hi = *(const v8h*)(p + 8);
  return cat8(lo, hi);
}

// B-fragment (32 K x 16 N) from N-major storage (row = N, contiguous K), ld in halfs.
// Lane lr holds column n0+lr; hl=0: K 0..15, hl=1: K 16..31.
DEV v16h load_b32x16(const _Float16* __restrict__ base, int n0, int k0, int ld, int lr, int hl) {
  const _Float16* p = base + (size_t)(n0 + lr) * ld + (k0 + 16 * hl);
  v8h lo = *(const v8h*)p;
  v8h hi = *(const v8h*)(p + 8);
  return cat8(lo, hi);
}

DEV v8f wmma_f16(v16h a, v16h b, v8f c) {
  return __builtin_amdgcn_wmma_f32_16x16x32_f16(false, a, false, b, (short)0, c, false, false);
}

DEV unsigned pk2(float a, float b) {
  v2h t; t[0] = (_Float16)a; t[1] = (_Float16)b;
  return __builtin_bit_cast(unsigned, t);
}

// ---------------- elementwise f32 -> f16 convert ----------------
__global__ void cvt_f16_kernel(const float* __restrict__ in, _Float16* __restrict__ out, int n) {
  int i = blockIdx.x * blockDim.x + threadIdx.x;
  if (i < n) out[i] = (_Float16)in[i];
}

// ---------------- x (bc, F, W) f32 -> xT (bc, W, F) f16, LDS-tiled ----------------
__global__ __launch_bounds__(256) void transpose_cvt_kernel(const float* __restrict__ x,
                                                            _Float16* __restrict__ xt) {
  __shared__ float tile[32][33];
  int bc = blockIdx.z;
  int w0 = blockIdx.x * 32, f0 = blockIdx.y * 32;
  const float* src = x + (size_t)bc * F_ * W_;
#pragma unroll
  for (int r = threadIdx.y; r < 32; r += 8)
    tile[r][threadIdx.x] = src[(size_t)(f0 + r) * W_ + w0 + threadIdx.x];
  __syncthreads();
  _Float16* dst = xt + (size_t)bc * W_ * F_;
#pragma unroll
  for (int r = threadIdx.y; r < 32; r += 8)
    dst[(size_t)(w0 + r) * F_ + f0 + threadIdx.x] = (_Float16)tile[threadIdx.x][r];
}

// ---------------- per-channel linear: Y[bc] = Wm[c] (FxF) @ X[bc] (FxW) + bias[c] ----------------
// Wm f16 row-major (C,F,F); Xt f16 w-major (BC, W, F); Y f32 (BC, F, W).
// Wave tile: 32 (f) x 64 (w). Block = 4 waves stacked along f -> 128 x 64 per block.
__global__ __launch_bounds__(128) void gemm_mcl_kernel(const _Float16* __restrict__ Wmat,
                                                       const _Float16* __restrict__ Xt,
                                                       const float* __restrict__ bias,
                                                       float* __restrict__ Y) {
  int lane = threadIdx.x & 31, wv = threadIdx.x >> 5;
  int lr = lane & 15, hl = lane >> 4;
  int bc = blockIdx.z, c = bc % C_;
  int f0 = blockIdx.y * 128 + wv * 32;
  int w0 = blockIdx.x * 64;
  const _Float16* A  = Wmat + (size_t)c * F_ * F_;
  const _Float16* Bm = Xt   + (size_t)bc * W_ * F_;

  v8f acc[2][4];
#pragma unroll
  for (int i = 0; i < 2; ++i)
#pragma unroll
    for (int t = 0; t < 4; ++t) acc[i][t] = (v8f)0.f;

  for (int kk = 0; kk < F_; kk += 32) {
    v16h a0 = load_a16x32(A, f0,      kk, F_, lr, hl);
    v16h a1 = load_a16x32(A, f0 + 16, kk, F_, lr, hl);
#pragma unroll
    for (int t = 0; t < 4; ++t) {
      v16h b = load_b32x16(Bm, w0 + 16 * t, kk, F_, lr, hl);
      acc[0][t] = wmma_f16(a0, b, acc[0][t]);
      acc[1][t] = wmma_f16(a1, b, acc[1][t]);
    }
  }

  float* out = Y + (size_t)bc * F_ * W_;
#pragma unroll
  for (int i = 0; i < 2; ++i)
#pragma unroll
    for (int j = 0; j < 8; ++j) {
      int f = f0 + 16 * i + j + 8 * hl;
      float bv = bias[c * F_ + f];
#pragma unroll
      for (int t = 0; t < 4; ++t)
        out[(size_t)f * W_ + w0 + 16 * t + lr] = acc[i][t][j] + bv;
    }
}

// ---------------- width conv, kw=7 pad=3, mixing channels ----------------
__global__ void conv7_kernel(const float* __restrict__ in, const float* __restrict__ cw,
                             const float* __restrict__ cb, float* __restrict__ out) {
  int idx = blockIdx.x * blockDim.x + threadIdx.x;
  if (idx >= B_ * C_ * F_ * W_) return;
  int w  = idx & (W_ - 1);
  int f  = (idx >> 10) & (F_ - 1);
  int co = (idx >> 19) & (C_ - 1);
  int b  = idx >> 22;
  float acc = cb[co];
#pragma unroll
  for (int ci = 0; ci < C_; ++ci) {
    const float* ip = in + ((size_t)(b * C_ + ci) * F_ + f) * W_;
#pragma unroll
    for (int t = 0; t < 7; ++t) {
      int ww = w + t - 3;
      if (ww >= 0 && ww < W_) acc += ip[ww] * cw[(co * C_ + ci) * 7 + t];
    }
  }
  out[idx] = acc;
}

// ---------------- 1x1 channel conv -> final output ----------------
__global__ void conv1_kernel(const float* __restrict__ in, const float* __restrict__ cw,
                             const float* __restrict__ cb, float* __restrict__ out) {
  int idx = blockIdx.x * blockDim.x + threadIdx.x;
  if (idx >= B_ * C_ * F_ * W_) return;
  int fw = idx & (F_ * W_ - 1);
  int co = (idx >> 19) & (C_ - 1);
  int b  = idx >> 22;
  float acc = cb[co];
#pragma unroll
  for (int ci = 0; ci < C_; ++ci)
    acc += in[(size_t)(b * C_ + ci) * F_ * W_ + fw] * cw[co * C_ + ci];
  out[idx] = acc;
}

// ---------------- RoPE + repack (b,c,f,w) f32 -> (bch, w, dh) f16; folds q scale ----------------
__global__ void rope_pack_kernel(const float* __restrict__ in, _Float16* __restrict__ out,
                                 float scale) {
  int idx = blockIdx.x * blockDim.x + threadIdx.x;       // (bch, w, p) p=0..31
  if (idx >= BCH_ * W_ * 32) return;
  int p   = idx & 31;
  int w   = (idx >> 5) & (W_ - 1);
  int bch = idx >> 15;
  int h   = bch & (H_ - 1);
  int bc  = bch >> 3;
  int f   = h * DH_ + 2 * p;
  const float* ip = in + ((size_t)bc * F_ + f) * W_ + w;
  float e0 = ip[0], e1 = ip[W_];
  // inv_freq = 10000^(-(2p)/64)
  float inv = __expf(-((float)(2 * p) / 64.f) * 9.210340371976184f);
  float ang = (float)w * inv;
  float s, c;
  __sincosf(ang, &s, &c);
  v2h pr;
  pr[0] = (_Float16)((e0 * c - e1 * s) * scale);
  pr[1] = (_Float16)((e1 * c + e0 * s) * scale);
  *(v2h*)(out + ((size_t)bch * W_ + w) * DH_ + 2 * p) = pr;
}

// ---------------- wave-level flash attention ----------------
// Q,K: (bch, w, dh) f16.  Vt: (bch, dh, w) f16.  Ot: (bc, w, F) f16 (w-major, heads merged).
// Computes S^T = K Q^T so softmax stats are per-lane; O^T = V^T P^T.
__global__ __launch_bounds__(128) void attn_kernel(const _Float16* __restrict__ Q,
                                                   const _Float16* __restrict__ K,
                                                   const _Float16* __restrict__ Vt,
                                                   _Float16* __restrict__ Ot) {
  int lane = threadIdx.x & 31, wv = threadIdx.x >> 5;
  int lr = lane & 15, hl = lane >> 4;
  int bch = blockIdx.y;
  int h   = bch & (H_ - 1);
  int bc  = bch >> 3;
  int q0  = blockIdx.x * 64 + wv * 16;

  const _Float16* Qb = Q  + (size_t)bch * W_ * DH_;
  const _Float16* Kb = K  + (size_t)bch * W_ * DH_;
  const _Float16* Vb = Vt + (size_t)bch * DH_ * W_;

  // Q^T fragments for this wave's 16 queries (K-dim = dh; two 32-chunks)
  v16h qb0 = load_b32x16(Qb, q0, 0,  DH_, lr, hl);
  v16h qb1 = load_b32x16(Qb, q0, 32, DH_, lr, hl);

  v8f o[4];
#pragma unroll
  for (int m = 0; m < 4; ++m) o[m] = (v8f)0.f;
  float mrun = -1e30f, lrun = 0.f;

  for (int kb = 0; kb < W_; kb += 64) {
    // S^T tiles: rows = keys, cols = queries (each lane holds one query's values)
    v8f s[4];
#pragma unroll
    for (int i = 0; i < 4; ++i) {
      v16h ka0 = load_a16x32(Kb, kb + 16 * i, 0,  DH_, lr, hl);
      v8f z = (v8f)0.f;
      s[i] = wmma_f16(ka0, qb0, z);
      v16h ka1 = load_a16x32(Kb, kb + 16 * i, 32, DH_, lr, hl);
      s[i] = wmma_f16(ka1, qb1, s[i]);
    }

    // online softmax: per-lane partial (32 of this block's 64 keys), one pair-swap combine
    float mloc = mrun;
#pragma unroll
    for (int i = 0; i < 4; ++i)
#pragma unroll
      for (int j = 0; j < 8; ++j) mloc = fmaxf(mloc, s[i][j]);
    float mnew = fmaxf(mloc, __shfl_xor(mloc, 16, 32));
    float rs = __expf(mrun - mnew);

    float pv[4][8];
    float ls = 0.f;
#pragma unroll
    for (int i = 0; i < 4; ++i)
#pragma unroll
      for (int j = 0; j < 8; ++j) {
        float e = __expf(s[i][j] - mnew);
        pv[i][j] = e;
        ls += e;
      }
    ls += __shfl_xor(ls, 16, 32);
    mrun = mnew;
    lrun = lrun * rs + ls;
#pragma unroll
    for (int m = 0; m < 4; ++m)
#pragma unroll
      for (int j = 0; j < 8; ++j) o[m][j] *= rs;

    // pack P (f32 pairs -> f16x2 dwords), per tile 4 dwords
    unsigned d[4][4];
#pragma unroll
    for (int i = 0; i < 4; ++i)
#pragma unroll
      for (int j2 = 0; j2 < 4; ++j2) d[i][j2] = pk2(pv[i][2 * j2], pv[i][2 * j2 + 1]);

    // build P^T B-fragments via lane<->lane+16 swaps, then O^T += V^T @ P^T
#pragma unroll
    for (int half = 0; half < 2; ++half) {
      unsigned fr[8];
#pragma unroll
      for (int q = 0; q < 4; ++q) {
        unsigned own  = d[2 * half + hl][q];
        unsigned send = d[2 * half + (1 - hl)][q];
        unsigned recv = (unsigned)__shfl_xor((int)send, 16, 32);
        fr[q]     = hl ? recv : own;
        fr[4 + q] = hl ? own  : recv;
      }
      v8u u;
#pragma unroll
      for (int q = 0; q < 8; ++q) u[q] = fr[q];
      v16h pb = __builtin_bit_cast(v16h, u);
#pragma unroll
      for (int m = 0; m < 4; ++m) {
        v16h va = load_a16x32(Vb, 16 * m, kb + 32 * half, W_, lr, hl);
        o[m] = wmma_f16(va, pb, o[m]);
      }
    }
  }

  float rl = 1.f / lrun;
  // store O^T directly in (bc, w, F) f16: lane's 8 dh values are contiguous
#pragma unroll
  for (int m = 0; m < 4; ++m) {
    v8h ph;
#pragma unroll
    for (int j = 0; j < 8; ++j) ph[j] = (_Float16)(o[m][j] * rl);
    *(v8h*)(Ot + ((size_t)(bc * W_ + q0 + lr)) * F_ + h * DH_ + 16 * m + 8 * hl) = ph;
  }
}

// ---------------- host-side orchestration ----------------
extern "C" void kernel_launch(void* const* d_in, const int* in_sizes, int n_in,
                              void* d_out, int out_size, void* d_ws, size_t ws_size,
                              hipStream_t stream) {
  (void)in_sizes; (void)n_in; (void)out_size; (void)ws_size;
  const float* x   = (const float*)d_in[0];
  const float* Wm[4] = { (const float*)d_in[1], (const float*)d_in[5],
                         (const float*)d_in[9], (const float*)d_in[13] };
  const float* bm[4] = { (const float*)d_in[2], (const float*)d_in[6],
                         (const float*)d_in[10], (const float*)d_in[14] };
  const float* cw[4] = { (const float*)d_in[3], (const float*)d_in[7],
                         (const float*)d_in[11], (const float*)d_in[15] };
  const float* cb[4] = { (const float*)d_in[4], (const float*)d_in[8],
                         (const float*)d_in[12], (const float*)d_in[16] };

  char* ws = (char*)d_ws;
  const size_t MB = 1ull << 20;
  _Float16* xT   = (_Float16*)(ws + 0);          // 16 MB: (BC, W, F) f16
  _Float16* w16[4] = { (_Float16*)(ws + 16*MB), (_Float16*)(ws + 20*MB),
                       (_Float16*)(ws + 24*MB), (_Float16*)(ws + 28*MB) }; // 4x4 MB
  float*    tmpA = (float*)(ws + 32*MB);         // 32 MB: gemm out (BC,F,W)
  float*    tmpB = (float*)(ws + 64*MB);         // 32 MB: conv out (BC,F,W)
  _Float16* Q16  = (_Float16*)(ws + 96*MB);      // 16 MB: (BCH, W, dh)
  _Float16* K16  = (_Float16*)(ws + 112*MB);     // 16 MB
  _Float16* V16  = (_Float16*)(ws + 128*MB);     // 16 MB: (BCH, dh, W)
  _Float16* OT   = (_Float16*)(ws + 144*MB);     // 16 MB: (BC, W, F)

  const int NW = C_ * F_ * F_;        // weight elems per projection
  const int N4 = B_ * C_ * F_ * W_;   // tensor elems
  const int NR = BCH_ * W_ * 32;      // rope threads
  const float qscale = 0.044194173824159216f;    // 1/sqrt(512)

  for (int p = 0; p < 4; ++p)
    cvt_f16_kernel<<<(NW + 255) / 256, 256, 0, stream>>>(Wm[p], w16[p], NW);

  transpose_cvt_kernel<<<dim3(W_ / 32, F_ / 32, BC_), dim3(32, 8), 0, stream>>>(x, xT);

  dim3 ggrid(W_ / 64, F_ / 128, BC_);
  // q / k / v projections
  for (int p = 0; p < 3; ++p) {
    gemm_mcl_kernel<<<ggrid, 128, 0, stream>>>(w16[p], xT, bm[p], tmpA);
    conv7_kernel<<<(N4 + 255) / 256, 256, 0, stream>>>(tmpA, cw[p], cb[p], tmpB);
    if (p == 0)
      rope_pack_kernel<<<(NR + 255) / 256, 256, 0, stream>>>(tmpB, Q16, qscale);
    else if (p == 1)
      rope_pack_kernel<<<(NR + 255) / 256, 256, 0, stream>>>(tmpB, K16, 1.0f);
    else
      cvt_f16_kernel<<<(N4 + 255) / 256, 256, 0, stream>>>(tmpB, V16, N4); // (b,c,f,w)==(bch,dh,w)
  }

  attn_kernel<<<dim3(W_ / 64, BCH_), 128, 0, stream>>>(Q16, K16, V16, OT);

  // output projection + 1x1 conv
  gemm_mcl_kernel<<<ggrid, 128, 0, stream>>>(w16[3], OT, bm[3], tmpA);
  conv1_kernel<<<(N4 + 255) / 256, 256, 0, stream>>>(tmpA, cw[3], cb[3], (float*)d_out);
}
